// HybridTwoTower_65506841198658
// MI455X (gfx1250) — matrix operand order repaired and checked by reading, not compiled, not run on hardware
//
#include <hip/hip_runtime.h>
#include <math.h>

typedef __attribute__((ext_vector_type(16))) __bf16        v16bf;
typedef __attribute__((ext_vector_type(8)))  float         v8f;
typedef __attribute__((ext_vector_type(8)))  unsigned int  v8u;

#define WAVES 4          // waves per block (wave32) -> 128 threads, 64 rows/block
#define TOW   128
#define HID   256

// ---- fp32 -> bf16 (round to nearest even) ----
static __device__ __forceinline__ unsigned short f2bf(float f) {
    unsigned int x = __builtin_bit_cast(unsigned int, f);
    unsigned int r = x + 0x7FFFu + ((x >> 16) & 1u);
    return (unsigned short)(r >> 16);
}
static __device__ __forceinline__ unsigned int pack2bf(float lo, float hi) {
    return (unsigned int)f2bf(lo) | ((unsigned int)f2bf(hi) << 16);
}

static __device__ __forceinline__ v8f wmma_bf16(v8u a, v8u b, v8f c) {
    return __builtin_amdgcn_wmma_f32_16x16x32_bf16(
        false, __builtin_bit_cast(v16bf, a),
        false, __builtin_bit_cast(v16bf, b),
        (short)0, c, false, false);
}

// =====================================================================
// Weight pre-pack: fp32 row-major W[K][N]  ->  bf16 WMMA B-fragments.
// Fragment (nt, ks) = 256 dwords, word index = L*8 + v so a lane loads
// its 8 dwords (2 x b128).  B layout (16-bit, 32x16 per ISA):
//   n = nt*16 + (L & 15),  k = ks*32 + (v>=4)*16 + (L>=16)*8 + (v&3)*2 + h
// =====================================================================
__global__ void pack_weights(const float* __restrict__ src,
                             unsigned int* __restrict__ dst,
                             int Ndim, int ksteps, int ntiles) {
    int gid = blockIdx.x * blockDim.x + threadIdx.x;
    int total = ntiles * ksteps * 256;
    if (gid >= total) return;
    int frag = gid >> 8;
    int w    = gid & 255;
    int L    = w >> 3;
    int v    = w & 7;
    int nt   = frag / ksteps;
    int ks   = frag % ksteps;
    int n    = nt * 16 + (L & 15);
    int k0   = ks * 32 + ((v >> 2) << 4) + ((L >> 4) << 3) + ((v & 3) << 1);
    float lo = src[(size_t)k0 * Ndim + n];
    float hi = src[(size_t)(k0 + 1) * Ndim + n];
    dst[gid] = pack2bf(lo, hi);
}

// =====================================================================
// Tower kernel: per wave, 16 rows through  relu(A @ W1 + b1) @ W2 + b2
// then row-wise L2 normalize.  MODE 0 = items (K=96), MODE 1 = users
// (K=192, with ragged history pooling against itemEmb).
//
// A layout in LDS (16-bit A 16x32 fragments, per ISA):
//   m = L & 15,  k = ks*32 + (v>=4)*16 + (L>=16)*8 + (v&3)*2 + h
// stored lane-major (word = L*8 + v) so reads are 2 x ds b128 per lane.
//
// Both GEMM loops keep several independent accumulator chains in
// flight so the scheduler can cover the WMMA->WMMA RAW hazard (5 slot)
// with real work instead of v_nop, and each A fragment is reused
// across 4 (layer 1) / 8 (layer 2) n-tiles.
// =====================================================================
template <int MODE, int KSTEPS>
__global__ __launch_bounds__(WAVES * 32)
void tower_kernel(const float* __restrict__ in0,      // items: item_feat   | users: gcn_user_emb
                  const float* __restrict__ gcnItem,  // items only
                  const float* __restrict__ Wgeo,     // items only (2x32)
                  const float* __restrict__ bgeo,     // items only (32)
                  const int*   __restrict__ hidx,     // users only (U x 10)
                  const int*   __restrict__ hlen,     // users only (U)
                  const float* __restrict__ itemEmb,  // users only (N x 128), from pass 1
                  const float* __restrict__ bias1,    // 256
                  const float* __restrict__ bias2,    // 128
                  const unsigned int* __restrict__ wf1,  // packed bf16 frags, layer 1
                  const unsigned int* __restrict__ wf2,  // packed bf16 frags, layer 2
                  float* __restrict__ out,            // nrows x 128
                  int nrows) {
    constexpr int ABUF = KSTEPS * 256;   // dwords of A fragments per wave
    constexpr int A2W  = 2048;           // dwords: 16x256 bf16 hid buffer
    __shared__ unsigned int lds[WAVES * (ABUF + A2W)];

    const int lane = threadIdx.x & 31;
    const int wave = threadIdx.x >> 5;
    unsigned int* Ab  = &lds[wave * (ABUF + A2W)];
    unsigned int* A2b = Ab + ABUF;
    const int r0 = (blockIdx.x * WAVES + wave) * 16;

    // ---------------- build A (16 x 32*KSTEPS) in fragment order -------------
    #pragma unroll 1
    for (int t = 0; t < ABUF / 32; ++t) {
        int gid = t * 32 + lane;
        int s  = gid >> 8;
        int w  = gid & 255;
        int Lp = w >> 3;
        int v  = w & 7;
        int m  = Lp & 15;
        int k  = s * 32 + (((v >> 2) << 4) | ((Lp >> 4) << 3) | ((v & 3) << 1));
        int row = r0 + m;
        bool ok = row < nrows;
        float lo = 0.f, hi = 0.f;
        if (MODE == 0) {
            if (k < 32) {                       // geo tower: relu(feat @ Wgeo + b)
                float f0 = 0.f, f1 = 0.f;
                if (ok) {
                    float2 f = ((const float2*)in0)[row];
                    f0 = f.x; f1 = f.y;
                }
                lo = fmaxf(f0 * Wgeo[k]     + f1 * Wgeo[32 + k]     + bgeo[k],     0.f);
                hi = fmaxf(f0 * Wgeo[k + 1] + f1 * Wgeo[32 + k + 1] + bgeo[k + 1], 0.f);
                if (!ok) { lo = 0.f; hi = 0.f; }
            } else {                            // gcn item embedding, cols k-32
                if (ok) {
                    float2 g = ((const float2*)gcnItem)[(size_t)row * 32 + ((k - 32) >> 1)];
                    lo = g.x; hi = g.y;
                }
            }
        } else {
            if (k < 64) {                       // gcn user embedding
                if (ok) {
                    float2 g = ((const float2*)in0)[(size_t)row * 32 + (k >> 1)];
                    lo = g.x; hi = g.y;
                }
            } else {                            // ragged weighted history pooling
                if (ok) {
                    int c2 = (k - 64) >> 1;
                    int Lh = hlen[row];
                    float denom = fmaxf((float)Lh - 1.0f, 1.0f);
                    float a0 = 0.f, a1 = 0.f, wsum = 0.f;
                    #pragma unroll
                    for (int j = 0; j < 10; ++j) {
                        if (j < Lh) {
                            float wj = 1.0f - 0.8f * (float)j / denom;
                            wsum += wj;
                            int id = hidx[row * 10 + j];
                            float2 e = ((const float2*)itemEmb)[(size_t)id * 64 + c2];
                            a0 += wj * e.x;
                            a1 += wj * e.y;
                        }
                    }
                    lo = a0 / wsum;
                    hi = a1 / wsum;
                }
            }
        }
        Ab[gid] = pack2bf(lo, hi);
    }
    __syncthreads();

    // ------------- layer 1: hid = relu(A @ W1 + b1), bf16 into A2 ------------
    // 4 independent accumulator chains per pass; A fragment shared by all 4.
    const int ncol = lane & 15;
    unsigned short* A2h = (unsigned short*)A2b;
    #pragma unroll 1
    for (int nt0 = 0; nt0 < 16; nt0 += 4) {
        v8f c[4];
        #pragma unroll
        for (int q = 0; q < 4; ++q) {
            float bv = bias1[(nt0 + q) * 16 + ncol];
            #pragma unroll
            for (int i = 0; i < 8; ++i) c[q][i] = bv;
        }
        #pragma unroll
        for (int ks = 0; ks < KSTEPS; ++ks) {
            v8u a = *(const v8u*)(Ab + ks * 256 + lane * 8);
            #pragma unroll
            for (int q = 0; q < 4; ++q) {
                v8u b = *(const v8u*)(wf1 + (((size_t)(nt0 + q) * KSTEPS + ks) << 8) + lane * 8);
                c[q] = wmma_bf16(a, b, c[q]);
            }
        }
        // relu + scatter bf16 into fragment-ordered A2 (k = output col)
        #pragma unroll
        for (int q = 0; q < 4; ++q) {
            int kcol = (nt0 + q) * 16 + ncol;
            int s2 = kcol >> 5, kp2 = kcol & 31, h = kcol & 1;
            int vv = (((kp2 >> 4) & 1) << 2) | ((kp2 >> 1) & 3);
            #pragma unroll
            for (int v = 0; v < 8; ++v) {
                int m   = v + ((lane >> 4) << 3);
                int Lp2 = (((kp2 >> 3) & 1) << 4) | m;
                A2h[(s2 * 256 + Lp2 * 8 + vv) * 2 + h] = f2bf(fmaxf(c[q][v], 0.f));
            }
        }
    }
    __syncthreads();

    // ------------- layer 2: out = A2 @ W2 + b2 -------------------------------
    // k-step outer: one A fragment feeds 8 independent accumulators.
    v8f acc[8];
    #pragma unroll
    for (int nt = 0; nt < 8; ++nt) {
        float bv = bias2[nt * 16 + ncol];
        #pragma unroll
        for (int i = 0; i < 8; ++i) acc[nt][i] = bv;
    }
    #pragma unroll 1
    for (int ks = 0; ks < 8; ++ks) {
        v8u a = *(const v8u*)(A2b + ks * 256 + lane * 8);
        #pragma unroll
        for (int nt = 0; nt < 8; ++nt) {
            v8u b = *(const v8u*)(wf2 + (((size_t)nt * 8 + ks) << 8) + lane * 8);
            acc[nt] = wmma_bf16(a, b, acc[nt]);
        }
    }

    // ---------------- row-wise L2 norm via cross-lane reduce ------------------
    // D layout: vgpr v holds row m = v + 8*(lane>=16), col n = nt*16 + (lane&15)
    float ps[8];
    #pragma unroll
    for (int v = 0; v < 8; ++v) {
        float s = 0.f;
        #pragma unroll
        for (int nt = 0; nt < 8; ++nt) s += acc[nt][v] * acc[nt][v];
        ps[v] = s;
    }
    #pragma unroll
    for (int v = 0; v < 8; ++v) {
        ps[v] += __shfl_xor(ps[v], 1, 32);
        ps[v] += __shfl_xor(ps[v], 2, 32);
        ps[v] += __shfl_xor(ps[v], 4, 32);
        ps[v] += __shfl_xor(ps[v], 8, 32);   // stays within 16-lane half
    }
    float rs[8];
    #pragma unroll
    for (int v = 0; v < 8; ++v) rs[v] = 1.0f / fmaxf(sqrtf(ps[v]), 1e-12f);

    #pragma unroll 1
    for (int nt = 0; nt < 8; ++nt) {
        #pragma unroll
        for (int v = 0; v < 8; ++v) {
            int row = r0 + v + ((lane >> 4) << 3);
            if (row < nrows)
                out[(size_t)row * TOW + nt * 16 + ncol] = acc[nt][v] * rs[v];
        }
    }
}

// ---- workspace layout (dword offsets) ----
#define OFF_I1 0            // 16 nt * 3 ks * 256 = 12288
#define OFF_I2 12288        //  8 nt * 8 ks * 256 = 16384
#define OFF_U1 28672        // 16 nt * 6 ks * 256 = 24576
#define OFF_U2 53248        //  8 nt * 8 ks * 256 = 16384  (total 278528 bytes)

extern "C" void kernel_launch(void* const* d_in, const int* in_sizes, int n_in,
                              void* d_out, int out_size, void* d_ws, size_t ws_size,
                              hipStream_t stream) {
    const float* item_feat = (const float*)d_in[0];
    const float* gcn_item  = (const float*)d_in[1];
    const float* gcn_user  = (const float*)d_in[2];
    const int*   hist_idx  = (const int*)  d_in[3];
    const int*   hist_len  = (const int*)  d_in[4];
    const float* W_geo     = (const float*)d_in[5];
    const float* b_geo     = (const float*)d_in[6];
    const float* W_i1      = (const float*)d_in[7];
    const float* b_i1      = (const float*)d_in[8];
    const float* W_i2      = (const float*)d_in[9];
    const float* b_i2      = (const float*)d_in[10];
    const float* W_u1      = (const float*)d_in[11];
    const float* b_u1      = (const float*)d_in[12];
    const float* W_u2      = (const float*)d_in[13];
    const float* b_u2      = (const float*)d_in[14];

    float* out = (float*)d_out;
    unsigned int* ws = (unsigned int*)d_ws;

    const int NI = in_sizes[0] / 2;   // N_ITEMS
    const int NU = in_sizes[4];       // N_USERS

    // 1) pack weights to bf16 WMMA fragments
    {
        int t1 = 16 * 3 * 256, t2 = 8 * 8 * 256, t3 = 16 * 6 * 256, t4 = 8 * 8 * 256;
        pack_weights<<<(t1 + 255) / 256, 256, 0, stream>>>(W_i1, ws + OFF_I1, 256, 3, 16);
        pack_weights<<<(t2 + 255) / 256, 256, 0, stream>>>(W_i2, ws + OFF_I2, 128, 8, 8);
        pack_weights<<<(t3 + 255) / 256, 256, 0, stream>>>(W_u1, ws + OFF_U1, 256, 6, 16);
        pack_weights<<<(t4 + 255) / 256, 256, 0, stream>>>(W_u2, ws + OFF_U2, 128, 8, 8);
    }

    // 2) item tower -> out[0 .. NI*128)
    tower_kernel<0, 3><<<(NI + WAVES * 16 - 1) / (WAVES * 16), WAVES * 32, 0, stream>>>(
        item_feat, gcn_item, W_geo, b_geo,
        nullptr, nullptr, nullptr,
        b_i1, b_i2, ws + OFF_I1, ws + OFF_I2,
        out, NI);

    // 3) user tower (reads item embeddings from out) -> out[NI*128 ..)
    tower_kernel<1, 6><<<(NU + WAVES * 16 - 1) / (WAVES * 16), WAVES * 32, 0, stream>>>(
        gcn_user, nullptr, nullptr, nullptr,
        hist_idx, hist_len, out,
        b_u1, b_u2, ws + OFF_U1, ws + OFF_U2,
        out + (size_t)NI * TOW, NU);
}
